// MutliHeadAttenation_39539468927195
// MI455X (gfx1250) — matrix operand order repaired and checked by reading, not compiled, not run on hardware
//
#include <hip/hip_runtime.h>
#include <math.h>

// Problem constants (B=2, S=2048, D=1024, H=16, DK=64)
#define BB  2
#define SS  2048
#define DD  1024
#define HH  16
#define DKK 64

typedef __attribute__((ext_vector_type(16))) __bf16          v16bf;
typedef __attribute__((ext_vector_type(8)))  float           v8f;
typedef __attribute__((ext_vector_type(16))) unsigned short  v16u;
typedef __attribute__((ext_vector_type(8)))  unsigned short  v8u;
typedef __attribute__((ext_vector_type(4)))  float           v4f;
typedef __attribute__((ext_vector_type(4)))  unsigned int    u32x4;
typedef __attribute__((ext_vector_type(8)))  unsigned int    u32x8;

__device__ __forceinline__ unsigned short bfbits(float f) {
  union { __bf16 b; unsigned short u; } x;
  x.b = (__bf16)f;               // native v_cvt_pk_bf16_f32 on gfx1250
  return x.u;
}

__device__ __forceinline__ v16bf cat8bf(v8u lo, v8u hi) {
  union { v16u u; v16bf b; } x;
  x.u = __builtin_shufflevector(lo, hi, 0,1,2,3,4,5,6,7,8,9,10,11,12,13,14,15);
  return x.b;
}

__device__ __forceinline__ v8f wmma_bf16(v16bf a, v16bf b, v8f c) {
  return __builtin_amdgcn_wmma_f32_16x16x32_bf16(
      false, a, false, b, (short)0, c, false, false);
}

// ---------------------------------------------------------------------------
// Tensor Data Mover: one descriptor-driven 2D tile DMA  global -> LDS.
// D# layout per CDNA5 ISA (2D: VADDR2/3 = NULL).
// pad_interval/pad_amount give padded, bank-conflict-free LDS rows.
// ---------------------------------------------------------------------------
__device__ __forceinline__ void tdm_load_2d(const void* gaddr, unsigned lds_off,
                                            unsigned tile_w, unsigned tile_h,
                                            unsigned row_stride_elems,
                                            unsigned pad_interval,
                                            unsigned pad_amount) {
  const unsigned long long ga = (unsigned long long)(size_t)gaddr;
  u32x4 g0;
  g0[0] = 1u;                                            // count=1 (valid user D#)
  g0[1] = lds_off;                                       // LDS byte address
  g0[2] = (unsigned)(ga & 0xFFFFFFFFu);                  // global_addr[31:0]
  g0[3] = (unsigned)((ga >> 32) & 0x01FFFFFFu)           // global_addr[56:32]
        | 0x80000000u;                                   // type=2 ("image")
  u32x8 g1;
  g1[0] = (1u << 16)                                     // data_size = 2 bytes
        | (1u << 20)                                     // pad_enable
        | (pad_interval << 22) | (pad_amount << 25);
  g1[1] = (tile_w & 0xFFFFu) << 16;                      // tensor_dim0 = tile_w
  g1[2] = (tile_w >> 16) | ((tile_h & 0xFFFFu) << 16);   // tensor_dim1 = tile_h
  g1[3] = (tile_h >> 16) | (tile_w << 16);               // tile_dim0
  g1[4] = tile_h & 0xFFFFu;                              // tile_dim1 (tile_dim2=0)
  g1[5] = row_stride_elems;                              // tensor_dim0_stride lo
  g1[6] = 0u;
  g1[7] = 0u;
  asm volatile("tensor_load_to_lds %0, %1" :: "s"(g0), "s"(g1) : "memory");
}

// ---------------------------------------------------------------------------
// Kernel 0: one-shot fp32 -> bf16 conversion (bandwidth-bound pre-pass).
// Removes all converts from the GEMM inner loops.
// ---------------------------------------------------------------------------
__global__ __launch_bounds__(256) void f32_to_bf16_kernel(
    const float* __restrict__ src, unsigned short* __restrict__ dst, int n)
{
  const int i = (blockIdx.x * 256 + threadIdx.x) * 8;
  if (i < n) {                       // n is a multiple of 8
    v4f f0 = *(const v4f*)(src + i);
    v4f f1 = *(const v4f*)(src + i + 4);
    v8u r;
#pragma unroll
    for (int j = 0; j < 4; ++j) { r[j] = bfbits(f0[j]); r[4 + j] = bfbits(f1[j]); }
    *(v8u*)(dst + i) = r;
  }
}

// ---------------------------------------------------------------------------
// Kernel 1: fused QKV projection (all-bf16 streams).  Y = X @ W^T + bias.
// One wave computes a 16x64 strip: 10 b128 loads + 4 WMMAs per K-step,
// zero converts in the loop.
//   z==0 -> Qb [B,H,S,DK];  z==1 -> Kb [B,H,S,DK];  z==2 -> Vt [B,H,DK,S]
// ---------------------------------------------------------------------------
__global__ __launch_bounds__(128) void qkv_proj_kernel(
    const unsigned short* __restrict__ xq, const unsigned short* __restrict__ xk,
    const unsigned short* __restrict__ xv,
    const unsigned short* __restrict__ wqb, const unsigned short* __restrict__ wkb,
    const unsigned short* __restrict__ wvb,
    const float* __restrict__ bq, const float* __restrict__ bk,
    const float* __restrict__ bv,
    unsigned short* __restrict__ Qb, unsigned short* __restrict__ Kb,
    unsigned short* __restrict__ Vt)
{
  const int lane  = threadIdx.x & 31;
  const int wid   = threadIdx.x >> 5;
  const int mtile = blockIdx.x;                    // 0..(B*S/16 - 1)
  const int nbase = (blockIdx.y * 4 + wid) * 64;   // 64-col strip per wave
  const int which = blockIdx.z;                    // 0=Q 1=K 2=V

  const unsigned short* x;  const unsigned short* w;  const float* bias;
  if (which == 0)      { x = xq; w = wqb; bias = bq; }
  else if (which == 1) { x = xk; w = wkb; bias = bk; }
  else                 { x = xv; w = wvb; bias = bv; }

  const int m    = lane & 15;
  const int half = lane >> 4;

  const unsigned short* xrow = x + (size_t)(mtile * 16 + m) * DD;  // A: row = M
  const unsigned short* wrow[4];
#pragma unroll
  for (int t = 0; t < 4; ++t)
    wrow[t] = w + (size_t)(nbase + t * 16 + m) * DD;               // B: lane = N

  v8f acc[4] = {v8f{}, v8f{}, v8f{}, v8f{}};
  for (int k0 = 0; k0 < DD; k0 += 32) {
    // issue all fragment loads up front so they pipeline
    v8u a0 = *(const v8u*)(xrow + k0 + half * 16);
    v8u a1 = *(const v8u*)(xrow + k0 + half * 16 + 8);
    v8u bl[4], bh[4];
#pragma unroll
    for (int t = 0; t < 4; ++t) {
      const unsigned short* p = wrow[t] + k0 + half * 16;
      bl[t] = *(const v8u*)(p);
      bh[t] = *(const v8u*)(p + 8);
    }
    v16bf a = cat8bf(a0, a1);
#pragma unroll
    for (int t = 0; t < 4; ++t)
      acc[t] = wmma_bf16(a, cat8bf(bl[t], bh[t]), acc[t]);
  }

#pragma unroll
  for (int t = 0; t < 4; ++t) {
    const int ncol = nbase + t * 16 + m;
    const float bval = bias[ncol];
    const int h = ncol >> 6;
    const int d = ncol & 63;
#pragma unroll
    for (int j = 0; j < 8; ++j) {
      const int r    = mtile * 16 + j + 8 * half;
      const int bidx = r >> 11;
      const int s    = r & 2047;
      const unsigned short val = bfbits(acc[t][j] + bval);
      if (which == 2) {
        Vt[(((size_t)bidx * HH + h) * DKK + d) * SS + s] = val;
      } else {
        unsigned short* dst = (which == 0) ? Qb : Kb;
        dst[(((size_t)bidx * HH + h) * SS + s) * DKK + d] = val;
      }
    }
  }
}

// ---------------------------------------------------------------------------
// Kernel 2: causal flash attention.
// Block = 4 waves = 64 queries of one (b,h).  K/V 32-key tiles are DMA'd into
// LDS by the Tensor Data Mover (wave 0 issues 2 descriptors per tile, double
// buffered, TENSORcnt-synchronized); all 4 waves compute from the shared tile.
// ---------------------------------------------------------------------------
__global__ __launch_bounds__(128) void attn_kernel(
    const unsigned short* __restrict__ Qb,
    const unsigned short* __restrict__ Kb,
    const unsigned short* __restrict__ Vt,
    unsigned short* __restrict__ Ao)
{
  // padded rows (written by TDM pad feature): K 144B, V 80B, P 80B
  __shared__ __align__(16) unsigned short Ksh[2][32][72];
  __shared__ __align__(16) unsigned short Vsh[2][64][40];
  __shared__ __align__(16) unsigned short Pt[4][16][40];

  const int tid   = threadIdx.x;
  const int lane  = tid & 31;
  const int wid   = tid >> 5;
  const int qtile = blockIdx.x * 4 + wid;     // 0..127
  const int h     = blockIdx.y;
  const int bidx  = blockIdx.z;
  const int qBase = qtile * 16;
  const int qBlockBase = blockIdx.x * 64;

  const size_t bh = (size_t)bidx * HH + h;
  const unsigned short* Qh = Qb + bh * (size_t)SS * DKK;
  const unsigned short* Kh = Kb + bh * (size_t)SS * DKK;
  const unsigned short* Vh = Vt + bh * (size_t)DKK * SS;

  const int m    = lane & 15;
  const int half = lane >> 4;

  auto issue_tile = [&](int kb_, int buf) {
    if (wid == 0) {
      const int keyBase_ = kb_ * 32;
      tdm_load_2d(Kh + (size_t)keyBase_ * DKK,
                  (unsigned)(size_t)&Ksh[buf][0][0],
                  /*tile_w=*/DKK, /*tile_h=*/32, /*stride=*/DKK,
                  /*pad_interval=*/4, /*pad_amount=*/3);
      tdm_load_2d(Vh + keyBase_,
                  (unsigned)(size_t)&Vsh[buf][0][0],
                  /*tile_w=*/32, /*tile_h=*/DKK, /*stride=*/SS,
                  /*pad_interval=*/3, /*pad_amount=*/3);
    }
  };

  const int myNkb = ((qBase + 15) >> 5) + 1;        // this wave's causal extent
  const int nkb   = ((qBlockBase + 63) >> 5) + 1;   // block's causal extent

  issue_tile(0, 0);   // prefetch first tile while we load Q

  v16bf aq0, aq1;
  {
    const unsigned short* qrow = Qh + (size_t)(qBase + m) * DKK;
    aq0 = cat8bf(*(const v8u*)(qrow + half * 16),
                 *(const v8u*)(qrow + half * 16 + 8));
    aq1 = cat8bf(*(const v8u*)(qrow + 32 + half * 16),
                 *(const v8u*)(qrow + 32 + half * 16 + 8));
  }

  float mrow[8], lrow[8];
#pragma unroll
  for (int j = 0; j < 8; ++j) { mrow[j] = -3.0e38f; lrow[j] = 0.0f; }
  v8f o[4] = {v8f{}, v8f{}, v8f{}, v8f{}};

  const float scale = 0.125f;                 // 1/sqrt(DK)
  unsigned short* P = &Pt[wid][0][0];

  for (int kb = 0; kb < nkb; ++kb) {
    const int cur = kb & 1;
    const bool pre = (kb + 1 < nkb);
    if (pre) issue_tile(kb + 1, cur ^ 1);     // overlap next DMA with compute
    if (wid == 0) {
      if (pre) __builtin_amdgcn_s_wait_tensorcnt((short)2);  // tile kb done
      else     __builtin_amdgcn_s_wait_tensorcnt((short)0);
    }
    __syncthreads();                           // tile kb visible to all waves

    if (kb < myNkb) {   // wave-uniform: EXEC stays all-ones for WMMA
      const int keyBase = kb * 32;

      // ---- scores: S = Q K^T (from LDS K tile) ----
      v8f s[2];
#pragma unroll
      for (int nt = 0; nt < 2; ++nt) {
        const unsigned short* krow = &Ksh[cur][nt * 16 + m][0];
        v16bf b0 = cat8bf(*(const v8u*)(krow + half * 16),
                          *(const v8u*)(krow + half * 16 + 8));
        v16bf b1 = cat8bf(*(const v8u*)(krow + 32 + half * 16),
                          *(const v8u*)(krow + 32 + half * 16 + 8));
        v8f sc = {};
        sc = wmma_bf16(aq0, b0, sc);
        sc = wmma_bf16(aq1, b1, sc);
        s[nt] = sc;
      }

      // ---- analytic causal mask + scale ----
#pragma unroll
      for (int nt = 0; nt < 2; ++nt) {
        const int key = keyBase + nt * 16 + m;
#pragma unroll
        for (int j = 0; j < 8; ++j) {
          const int qr = qBase + j + 8 * half;
          const float val = s[nt][j] * scale;
          s[nt][j] = (key > qr) ? -1.0e30f : val;
        }
      }

      // ---- online softmax: row stats over the 16 lanes of each half ----
#pragma unroll
      for (int j = 0; j < 8; ++j) {
        float vmax = fmaxf(s[0][j], s[1][j]);
#pragma unroll
        for (int off = 1; off < 16; off <<= 1)
          vmax = fmaxf(vmax, __shfl_xor(vmax, off, 32));
        const float mnew = fmaxf(mrow[j], vmax);
        const float corr = __expf(mrow[j] - mnew);
        mrow[j] = mnew;
        const float p0 = __expf(s[0][j] - mnew);
        const float p1 = __expf(s[1][j] - mnew);
        s[0][j] = p0; s[1][j] = p1;
        float rs = p0 + p1;
#pragma unroll
        for (int off = 1; off < 16; off <<= 1)
          rs += __shfl_xor(rs, off, 32);
        lrow[j] = lrow[j] * corr + rs;
#pragma unroll
        for (int t = 0; t < 4; ++t) o[t][j] *= corr;
      }

      // ---- P: C-layout -> A-layout via per-wave LDS tile ----
#pragma unroll
      for (int j = 0; j < 8; ++j) {
        P[(j + 8 * half) * 40 + m]      = bfbits(s[0][j]);
        P[(j + 8 * half) * 40 + 16 + m] = bfbits(s[1][j]);
      }
      asm volatile("s_wait_dscnt 0x0" ::: "memory");

      const unsigned short* prow = P + m * 40;
      v16bf ap = cat8bf(*(const v8u*)(prow + half * 16),
                        *(const v8u*)(prow + half * 16 + 8));

      // ---- O += P @ V (from LDS V tile, key-contiguous rows) ----
#pragma unroll
      for (int t = 0; t < 4; ++t) {
        const unsigned short* vrow = &Vsh[cur][t * 16 + m][half * 16];
        v16bf bv_ = cat8bf(*(const v8u*)(vrow), *(const v8u*)(vrow + 8));
        o[t] = wmma_bf16(ap, bv_, o[t]);
      }
    }
    __syncthreads();    // protect buffer reused by next iteration's DMA
  }

  // ---- normalize, merge heads: Ao [B,S,D] bf16 ----
#pragma unroll
  for (int j = 0; j < 8; ++j) {
    const float inv = 1.0f / lrow[j];
    const int srow = qBase + j + 8 * half;
    const size_t base = ((size_t)bidx * SS + srow) * DD + (size_t)h * DKK;
#pragma unroll
    for (int t = 0; t < 4; ++t)
      Ao[base + t * 16 + m] = bfbits(o[t][j] * inv);
  }
}

// ---------------------------------------------------------------------------
// Kernel 3: output projection (all-bf16 streams).  out = Ao @ Wo^T + bo (fp32)
// ---------------------------------------------------------------------------
__global__ __launch_bounds__(128) void out_proj_kernel(
    const unsigned short* __restrict__ Ao, const unsigned short* __restrict__ wob,
    const float* __restrict__ bo, float* __restrict__ out)
{
  const int lane  = threadIdx.x & 31;
  const int wid   = threadIdx.x >> 5;
  const int mtile = blockIdx.x;
  const int nbase = (blockIdx.y * 4 + wid) * 64;

  const int m    = lane & 15;
  const int half = lane >> 4;

  const unsigned short* arow = Ao + (size_t)(mtile * 16 + m) * DD;
  const unsigned short* wrow[4];
#pragma unroll
  for (int t = 0; t < 4; ++t)
    wrow[t] = wob + (size_t)(nbase + t * 16 + m) * DD;

  v8f acc[4] = {v8f{}, v8f{}, v8f{}, v8f{}};
  for (int k0 = 0; k0 < DD; k0 += 32) {
    v8u a0 = *(const v8u*)(arow + k0 + half * 16);
    v8u a1 = *(const v8u*)(arow + k0 + half * 16 + 8);
    v8u bl[4], bh[4];
#pragma unroll
    for (int t = 0; t < 4; ++t) {
      const unsigned short* p = wrow[t] + k0 + half * 16;
      bl[t] = *(const v8u*)(p);
      bh[t] = *(const v8u*)(p + 8);
    }
    v16bf a = cat8bf(a0, a1);
#pragma unroll
    for (int t = 0; t < 4; ++t)
      acc[t] = wmma_bf16(a, cat8bf(bl[t], bh[t]), acc[t]);
  }

#pragma unroll
  for (int t = 0; t < 4; ++t) {
    const int ncol = nbase + t * 16 + m;
    const float bval = bo[ncol];
#pragma unroll
    for (int j = 0; j < 8; ++j) {
      const int r = mtile * 16 + j + 8 * half;
      out[(size_t)r * DD + ncol] = acc[t][j] + bval;
    }
  }
}

// ---------------------------------------------------------------------------
extern "C" void kernel_launch(void* const* d_in, const int* in_sizes, int n_in,
                              void* d_out, int out_size, void* d_ws, size_t ws_size,
                              hipStream_t stream) {
  const float* q  = (const float*)d_in[0];
  const float* k  = (const float*)d_in[1];
  const float* v  = (const float*)d_in[2];
  // d_in[3] = mask (int32 tril) -- applied analytically, never read
  const float* wq = (const float*)d_in[4];
  const float* bq = (const float*)d_in[5];
  const float* wk = (const float*)d_in[6];
  const float* bk = (const float*)d_in[7];
  const float* wv = (const float*)d_in[8];
  const float* bv = (const float*)d_in[9];
  const float* wo = (const float*)d_in[10];
  const float* bo = (const float*)d_in[11];
  float* out = (float*)d_out;

  const size_t nAct = (size_t)BB * SS * DD;      // 4M elements
  const size_t nW   = (size_t)DD * DD;           // 1M elements
  unsigned short* Qb  = (unsigned short*)d_ws;   // bf16 buffers in workspace
  unsigned short* Kb  = Qb  + nAct;
  unsigned short* Vt  = Kb  + nAct;
  unsigned short* Ao  = Vt  + nAct;
  unsigned short* xqb = Ao  + nAct;
  unsigned short* xkb = xqb + nAct;
  unsigned short* xvb = xkb + nAct;
  unsigned short* wqb = xvb + nAct;
  unsigned short* wkb = wqb + nW;
  unsigned short* wvb = wkb + nW;
  unsigned short* wob = wvb + nW;

  dim3 blk(128);
  // pre-pass: one-shot fp32->bf16 (bandwidth-bound)
  const int actBlocks = (int)(nAct / 8 / 256);
  const int wBlocks   = (int)(nW / 8 / 256);
  f32_to_bf16_kernel<<<actBlocks, 256, 0, stream>>>(q,  xqb, (int)nAct);
  f32_to_bf16_kernel<<<actBlocks, 256, 0, stream>>>(k,  xkb, (int)nAct);
  f32_to_bf16_kernel<<<actBlocks, 256, 0, stream>>>(v,  xvb, (int)nAct);
  f32_to_bf16_kernel<<<wBlocks,   256, 0, stream>>>(wq, wqb, (int)nW);
  f32_to_bf16_kernel<<<wBlocks,   256, 0, stream>>>(wk, wkb, (int)nW);
  f32_to_bf16_kernel<<<wBlocks,   256, 0, stream>>>(wv, wvb, (int)nW);
  f32_to_bf16_kernel<<<wBlocks,   256, 0, stream>>>(wo, wob, (int)nW);

  qkv_proj_kernel<<<dim3(BB * SS / 16, DD / 256, 3), blk, 0, stream>>>(
      xqb, xkb, xvb, wqb, wkb, wvb, bq, bk, bv, Qb, Kb, Vt);
  attn_kernel<<<dim3(SS / 64, HH, BB), blk, 0, stream>>>(Qb, Kb, Vt, Ao);
  out_proj_kernel<<<dim3(BB * SS / 16, DD / 256), blk, 0, stream>>>(Ao, wob, bo, out);
}
